// FD_VAE_18348100289076
// MI455X (gfx1250) — matrix operand order repaired and checked by reading, not compiled, not run on hardware
//
#include <hip/hip_runtime.h>
#include <hip/hip_bf16.h>
#include <math.h>

// ---------------------------------------------------------------------------
// Types for CDNA5 WMMA (wave32)
// ---------------------------------------------------------------------------
typedef __attribute__((ext_vector_type(8)))  float          v8f;
typedef __attribute__((ext_vector_type(16))) __bf16         v16bf;
typedef __attribute__((ext_vector_type(8)))  __bf16         v8bf;
typedef __attribute__((ext_vector_type(8)))  unsigned short us8;

union Frag16 {
    v8bf  h[2];   // two 16B halves
    v16bf v;      // 16 bf16 = 8 VGPRs
};

// hardware bf16 convert (RNE) -- lowers to v_cvt_*bf16_f32 on gfx1250
__device__ __forceinline__ __bf16 f2bf(float f) { return (__bf16)f; }

__device__ __forceinline__ float eluf(float v)      { return v > 0.f ? v : expm1f(v); }
__device__ __forceinline__ float softplusf(float v) { return v > 20.f ? v : log1pf(expf(v)); }
__device__ __forceinline__ float sigmoidf(float v)  { return 1.f / (1.f + expf(-v)); }

#define F_GNN 128
#define F_IN  512
#define HIDN  64
#define ZDIM  64

// ---------------------------------------------------------------------------
// 0) Wg (f32, [512][128]) -> bf16 TRANSPOSED wghT [128 cols][512 k]
//    (one-time; makes per-tile B staging a contiguous b128 copy)
// ---------------------------------------------------------------------------
__global__ void k_convert_wgT(const float* __restrict__ Wg, __bf16* __restrict__ wghT, int n) {
    int i = blockIdx.x * blockDim.x + threadIdx.x;
    if (i < n) {
        int c = i >> 9;          // col 0..127
        int k = i & 511;         // k   0..511
        wghT[i] = f2bf(Wg[(size_t)k * F_GNN + c]);
    }
}

// ---------------------------------------------------------------------------
// 1) degree (self-loop contributes 1), then dinv = rsqrt(deg)
// ---------------------------------------------------------------------------
__global__ void k_init_deg(float* __restrict__ deg, int n) {
    int i = blockIdx.x * blockDim.x + threadIdx.x;
    if (i < n) deg[i] = 1.0f;                     // self loop
}
__global__ void k_deg_edges(const int* __restrict__ dst, float* __restrict__ deg, int E) {
    int i = blockIdx.x * blockDim.x + threadIdx.x;
    if (i < E) atomicAdd(&deg[dst[i]], 1.0f);
}
__global__ void k_dinv(float* __restrict__ deg, int n) {
    int i = blockIdx.x * blockDim.x + threadIdx.x;
    if (i < n) deg[i] = rsqrtf(deg[i]);           // deg >= 1 always
}

// ---------------------------------------------------------------------------
// 2) xw = x @ Wg  --- bf16 WMMA GEMM, [N,512] @ [512,128] -> [N,128] (f32)
//    block = 256 threads (8 waves); tile = 128 rows x 128 cols;
//    wave w -> cols [16w,16w+16), 8 row-tiles of 16; LDS double-buffered.
// ---------------------------------------------------------------------------
#define TILE_M 128
#define KC     32
#define NSTEP  (F_IN / KC)   // 16
#define LDSS   40            // bf16 pitch: 80B -> conflict-free frag reads, 16B aligned

__global__ __launch_bounds__(256) void k_gemm_xw(const float* __restrict__ x,
                                                 const __bf16* __restrict__ wghT,
                                                 float* __restrict__ xw, int nrows) {
    __shared__ __bf16 sA[2][TILE_M][LDSS];   // 128 x 32 bf16 per buffer
    __shared__ __bf16 sB[2][F_GNN][LDSS];    // [col][k] per buffer

    const int tid  = threadIdx.x;
    const int wave = tid >> 5;
    const int lane = tid & 31;
    const int rowBase = blockIdx.x * TILE_M;

    // staging coords: each thread moves 16 A elems and 16 B elems per K-step
    const int aR = tid >> 1;              // A row within tile   0..127
    const int aC = (tid & 1) * 16;        // A k-offset          0 | 16
    const int bC = tid >> 1;              // B col               0..127
    const int bK = (tid & 1) * 16;        // B k-offset          0 | 16

    const bool aValid = (rowBase + aR) < nrows;
    const float* aPtr = x + (size_t)(rowBase + aR) * F_IN + aC;
    const __bf16* bPtr = wghT + (size_t)bC * F_IN + bK;

    v8f acc[8] = {};

    float4 aReg[4];
    v8bf   bReg[2];

    auto load_step = [&](int k0) {
        if (aValid) {
            aReg[0] = *(const float4*)(aPtr + k0);
            aReg[1] = *(const float4*)(aPtr + k0 + 4);
            aReg[2] = *(const float4*)(aPtr + k0 + 8);
            aReg[3] = *(const float4*)(aPtr + k0 + 12);
        } else {
            aReg[0] = aReg[1] = aReg[2] = aReg[3] = make_float4(0.f, 0.f, 0.f, 0.f);
        }
        bReg[0] = *(const v8bf*)(bPtr + k0);
        bReg[1] = *(const v8bf*)(bPtr + k0 + 8);
    };
    auto store_step = [&](int buf) {
        // pack 16 floats -> 16 bf16 via HW converts
        const float* af = (const float*)aReg;
        v8bf p0, p1;
        #pragma unroll
        for (int i = 0; i < 8; ++i) p0[i] = f2bf(af[i]);
        #pragma unroll
        for (int i = 0; i < 8; ++i) p1[i] = f2bf(af[8 + i]);
        *(v8bf*)&sA[buf][aR][aC]     = p0;
        *(v8bf*)&sA[buf][aR][aC + 8] = p1;
        *(v8bf*)&sB[buf][bC][bK]     = bReg[0];
        *(v8bf*)&sB[buf][bC][bK + 8] = bReg[1];
    };

    // ---- prologue ----
    load_step(0);
    store_step(0);
    __syncthreads();

    const int nCol  = wave * 16 + (lane & 15);
    const int kbB   = (lane < 16) ? 0 : 8;
    const int aRow  = lane & 15;
    const int abase = (lane < 16) ? 0 : 8;

    for (int step = 0; step < NSTEP; ++step) {
        const int buf = step & 1;
        if (step + 1 < NSTEP) load_step((step + 1) * KC);                 // overlap w/ compute
        if (step + 2 < NSTEP && aValid)                                   // HBM hint 2 ahead
            __builtin_prefetch(aPtr + (step + 2) * KC, 0, 1);

        // B fragment: lane holds col nCol; lanes 0-15: K {0..7,16..23}, lanes 16-31: K {8..15,24..31}
        Frag16 bf;
        bf.h[0] = *(const v8bf*)&sB[buf][nCol][kbB];
        bf.h[1] = *(const v8bf*)&sB[buf][nCol][kbB + 16];

        // A fragments + WMMA: lanes 0-15 K {0..7,16..23}; lanes 16-31 K {8..15,24..31}
        #pragma unroll
        for (int rt = 0; rt < 8; ++rt) {
            Frag16 af;
            af.h[0] = *(const v8bf*)&sA[buf][rt * 16 + aRow][abase];
            af.h[1] = *(const v8bf*)&sA[buf][rt * 16 + aRow][abase + 16];
            acc[rt] = __builtin_amdgcn_wmma_f32_16x16x32_bf16(
                false, af.v, false, bf.v, (short)0, acc[rt], false, false);
        }

        if (step + 1 < NSTEP) {
            store_step(buf ^ 1);
            __syncthreads();
        }
    }

    // ---- store: C/D layout: VGPR j -> row m=j (lanes 0-15) / m=8+j (lanes 16-31) ----
    const int col  = wave * 16 + (lane & 15);
    const int rsel = (lane < 16) ? 0 : 8;
    if (rowBase + TILE_M <= nrows) {                         // fast path: no guards
        #pragma unroll
        for (int rt = 0; rt < 8; ++rt) {
            #pragma unroll
            for (int j = 0; j < 8; ++j) {
                const int row = rowBase + rt * 16 + rsel + j;
                xw[(size_t)row * F_GNN + col] = acc[rt][j];
            }
        }
    } else {
        #pragma unroll
        for (int rt = 0; rt < 8; ++rt) {
            #pragma unroll
            for (int j = 0; j < 8; ++j) {
                const int row = rowBase + rt * 16 + rsel + j;
                if (row < nrows) xw[(size_t)row * F_GNN + col] = acc[rt][j];
            }
        }
    }
}

// ---------------------------------------------------------------------------
// 3) agg init with self-loop term: agg[n,:] = xw[n,:] * dinv[n]^2
// ---------------------------------------------------------------------------
__global__ void k_init_agg(const float* __restrict__ xw, const float* __restrict__ dinv,
                           float* __restrict__ agg, int n) {
    int tid = blockIdx.x * blockDim.x + threadIdx.x;     // n * 32 threads
    if (tid >= n * 32) return;
    int node = tid >> 5;
    int c4   = (tid & 31) << 2;
    float s  = dinv[node] * dinv[node];
    const float4 v = *(const float4*)(xw + (size_t)node * F_GNN + c4);
    float4 r = make_float4(v.x * s, v.y * s, v.z * s, v.w * s);
    *(float4*)(agg + (size_t)node * F_GNN + c4) = r;
}

// ---------------------------------------------------------------------------
// 4) edge scatter: agg[dst] += xw[src] * dinv[src]*dinv[dst]  (L2-resident)
// ---------------------------------------------------------------------------
__global__ void k_edge_scatter(const int* __restrict__ src, const int* __restrict__ dst,
                               const float* __restrict__ dinv, const float* __restrict__ xw,
                               float* __restrict__ agg, int E) {
    int tid = blockIdx.x * blockDim.x + threadIdx.x;     // E * 32 threads
    if (tid >= E * 32) return;
    int e  = tid >> 5;
    int c4 = (tid & 31) << 2;
    int s  = src[e], d = dst[e];
    float norm = dinv[s] * dinv[d];
    const float4 v = *(const float4*)(xw + (size_t)s * F_GNN + c4);
    float* a = agg + (size_t)d * F_GNN + c4;
    atomicAdd(a + 0, v.x * norm);
    atomicAdd(a + 1, v.y * norm);
    atomicAdd(a + 2, v.z * norm);
    atomicAdd(a + 3, v.w * norm);
}

// ---------------------------------------------------------------------------
// 5) zero fill pool buffers
// ---------------------------------------------------------------------------
__global__ void k_zero(float* __restrict__ p, int n) {
    int i = blockIdx.x * blockDim.x + threadIdx.x;
    if (i < n) p[i] = 0.0f;
}

// ---------------------------------------------------------------------------
// 6) node_x = relu(agg + bg); accumulate mean-sum / max / count per graph
//    (post-ReLU values >= 0, so int-compare atomicMax == float max, init 0)
// ---------------------------------------------------------------------------
__global__ void k_node_pool(const float* __restrict__ agg, const float* __restrict__ bg,
                            const int* __restrict__ batch, float* __restrict__ msum,
                            float* __restrict__ mmax, float* __restrict__ cnt, int n) {
    int tid = blockIdx.x * blockDim.x + threadIdx.x;     // n * 128 threads
    if (tid >= n * F_GNN) return;
    int node = tid >> 7;
    int f    = tid & 127;
    float v  = agg[(size_t)node * F_GNN + f] + bg[f];
    v = v > 0.f ? v : 0.f;
    int g = batch[node];
    atomicAdd(&msum[(size_t)g * F_GNN + f], v);
    atomicMax((int*)&mmax[(size_t)g * F_GNN + f], __float_as_int(v));
    if (f == 0) atomicAdd(&cnt[g], 1.0f);
}

// ---------------------------------------------------------------------------
// 7) graph_x = [mean | max]  ([G, 256])
// ---------------------------------------------------------------------------
__global__ void k_pool_finalize(const float* __restrict__ msum, const float* __restrict__ mmax,
                                const float* __restrict__ cnt, float* __restrict__ gx, int G) {
    int tid = blockIdx.x * blockDim.x + threadIdx.x;     // G * 128
    if (tid >= G * F_GNN) return;
    int g = tid >> 7;
    int f = tid & 127;
    float c = fmaxf(cnt[g], 1.0f);
    gx[(size_t)g * 256 + f]       = msum[(size_t)g * F_GNN + f] / c;
    gx[(size_t)g * 256 + 128 + f] = mmax[(size_t)g * F_GNN + f];
}

// ---------------------------------------------------------------------------
// 8) generic tiny linear layer with activation  (act: 0=none, 1=elu, 2=tanh)
// ---------------------------------------------------------------------------
__global__ void k_linact(const float* __restrict__ in, const float* __restrict__ W,
                         const float* __restrict__ b, float* __restrict__ out,
                         int G, int K, int Fo, int act) {
    int tid = blockIdx.x * blockDim.x + threadIdx.x;
    if (tid >= G * Fo) return;
    int g = tid / Fo, h = tid % Fo;
    const float* gi = in + (size_t)g * K;
    float acc = b[h];
    for (int k = 0; k < K; ++k) acc = fmaf(gi[k], W[(size_t)k * Fo + h], acc);
    if (act == 1) acc = eluf(acc);
    else if (act == 2) acc = tanhf(acc);
    out[(size_t)g * Fo + h] = acc;
}

// ---------------------------------------------------------------------------
// 9) encoder head: ml = t2 @ We3 + be3; mu/stddev/z (reparameterize)
// ---------------------------------------------------------------------------
__global__ void k_enc3_reparam(const float* __restrict__ t2, const float* __restrict__ We3,
                               const float* __restrict__ be3, const float* __restrict__ eps,
                               float* __restrict__ out_mu, float* __restrict__ out_sd,
                               float* __restrict__ z, int G) {
    int tid = blockIdx.x * blockDim.x + threadIdx.x;     // G * 64
    if (tid >= G * ZDIM) return;
    int g = tid >> 6, j = tid & 63;
    const float* gi = t2 + (size_t)g * HIDN;
    float mu = be3[j], lv = be3[j + ZDIM];
    for (int k = 0; k < HIDN; ++k) {
        float v = gi[k];
        mu = fmaf(v, We3[(size_t)k * (2 * ZDIM) + j], mu);
        lv = fmaf(v, We3[(size_t)k * (2 * ZDIM) + j + ZDIM], lv);
    }
    float sd = 1e-6f + softplusf(lv);
    out_mu[(size_t)g * ZDIM + j] = mu;
    out_sd[(size_t)g * ZDIM + j] = sd;
    z[(size_t)g * ZDIM + j]      = mu + eps[(size_t)g * ZDIM + j] * sd;
}

// ---------------------------------------------------------------------------
// 10) decoder head: y = clip(sigmoid(d2 @ Wd3 + bd3))
// ---------------------------------------------------------------------------
__global__ void k_dec3(const float* __restrict__ d2, const float* __restrict__ Wd3,
                       const float* __restrict__ bd3, float* __restrict__ y, int G) {
    int tid = blockIdx.x * blockDim.x + threadIdx.x;     // G * 128
    if (tid >= G * F_GNN) return;
    int g = tid >> 7, j = tid & 127;
    const float* gi = d2 + (size_t)g * HIDN;
    float acc = bd3[j];
    for (int k = 0; k < HIDN; ++k) acc = fmaf(gi[k], Wd3[(size_t)k * F_GNN + j], acc);
    float v = sigmoidf(acc);
    v = fminf(fmaxf(v, 1e-8f), 1.0f - 1e-8f);
    y[(size_t)g * F_GNN + j] = v;
}

// ---------------------------------------------------------------------------
// launch
// ---------------------------------------------------------------------------
extern "C" void kernel_launch(void* const* d_in, const int* in_sizes, int n_in,
                              void* d_out, int out_size, void* d_ws, size_t ws_size,
                              hipStream_t stream) {
    const float* x     = (const float*)d_in[0];
    const int*   ei    = (const int*)d_in[1];
    const int*   batch = (const int*)d_in[2];
    const float* eps   = (const float*)d_in[3];
    const float* Wg  = (const float*)d_in[4];  const float* bg  = (const float*)d_in[5];
    const float* We1 = (const float*)d_in[6];  const float* be1 = (const float*)d_in[7];
    const float* We2 = (const float*)d_in[8];  const float* be2 = (const float*)d_in[9];
    const float* We3 = (const float*)d_in[10]; const float* be3 = (const float*)d_in[11];
    const float* Wd1 = (const float*)d_in[12]; const float* bd1 = (const float*)d_in[13];
    const float* Wd2 = (const float*)d_in[14]; const float* bd2 = (const float*)d_in[15];
    const float* Wd3 = (const float*)d_in[16]; const float* bd3 = (const float*)d_in[17];

    const int N = in_sizes[0] / F_IN;       // 100000
    const int E = in_sizes[1] / 2;          // 1600000
    const int G = in_sizes[3] / ZDIM;       // 512

    const int* src = ei;
    const int* dst = ei + E;

    // workspace carve-out
    char*  base = (char*)d_ws;
    size_t off  = 0;
    auto carve = [&](size_t bytes) -> char* {
        char* p = base + off;
        off = (off + bytes + 255) & ~(size_t)255;
        return p;
    };
    __bf16* wghT = (__bf16*)carve((size_t)F_IN * F_GNN * 2);  // [128][512] bf16
    float* xw   = (float*)carve((size_t)N * F_GNN * 4);
    float* agg  = (float*)carve((size_t)N * F_GNN * 4);
    float* dinv = (float*)carve((size_t)N * 4);
    float* pool = (float*)carve((size_t)(G * F_GNN * 2 + G) * 4);   // msum|mmax|cnt
    float* msum = pool;
    float* mmax = pool + (size_t)G * F_GNN;
    float* cnt  = pool + (size_t)G * F_GNN * 2;
    float* gx   = (float*)carve((size_t)G * 256 * 4);
    float* t1   = (float*)carve((size_t)G * HIDN * 4);
    float* t2   = (float*)carve((size_t)G * HIDN * 4);
    float* z    = (float*)carve((size_t)G * ZDIM * 4);
    float* dd1  = (float*)carve((size_t)G * HIDN * 4);
    float* dd2  = (float*)carve((size_t)G * HIDN * 4);

    float* out_mu = (float*)d_out;                       // [G, 64]
    float* out_sd = out_mu + (size_t)G * ZDIM;           // [G, 64]
    float* out_y  = out_sd + (size_t)G * ZDIM;           // [G, 128]

    const int BT = 256;
    auto blocks = [&](long long n) { return (int)((n + BT - 1) / BT); };

    // weight conversion (transposed) + degree normalization
    k_convert_wgT<<<blocks(F_IN * F_GNN), BT, 0, stream>>>(Wg, wghT, F_IN * F_GNN);
    k_init_deg  <<<blocks(N), BT, 0, stream>>>(dinv, N);
    k_deg_edges <<<blocks(E), BT, 0, stream>>>(dst, dinv, E);
    k_dinv      <<<blocks(N), BT, 0, stream>>>(dinv, N);

    // big GEMM via WMMA (double-buffered LDS pipeline)
    k_gemm_xw<<<(N + TILE_M - 1) / TILE_M, 256, 0, stream>>>(x, wghT, xw, N);

    // aggregation
    k_init_agg    <<<blocks((long long)N * 32), BT, 0, stream>>>(xw, dinv, agg, N);
    k_edge_scatter<<<blocks((long long)E * 32), BT, 0, stream>>>(src, dst, dinv, xw, agg, E);

    // pooling
    k_zero         <<<blocks(G * F_GNN * 2 + G), BT, 0, stream>>>(pool, G * F_GNN * 2 + G);
    k_node_pool    <<<blocks((long long)N * F_GNN), BT, 0, stream>>>(agg, bg, batch, msum, mmax, cnt, N);
    k_pool_finalize<<<blocks(G * F_GNN), BT, 0, stream>>>(msum, mmax, cnt, gx, G);

    // encoder MLP
    k_linact<<<blocks(G * HIDN), BT, 0, stream>>>(gx, We1, be1, t1, G, 2 * F_GNN, HIDN, 1); // elu
    k_linact<<<blocks(G * HIDN), BT, 0, stream>>>(t1, We2, be2, t2, G, HIDN, HIDN, 2);      // tanh
    k_enc3_reparam<<<blocks(G * ZDIM), BT, 0, stream>>>(t2, We3, be3, eps, out_mu, out_sd, z, G);

    // decoder MLP
    k_linact<<<blocks(G * HIDN), BT, 0, stream>>>(z,   Wd1, bd1, dd1, G, ZDIM, HIDN, 2);    // tanh
    k_linact<<<blocks(G * HIDN), BT, 0, stream>>>(dd1, Wd2, bd2, dd2, G, HIDN, HIDN, 1);    // elu
    k_dec3  <<<blocks(G * F_GNN), BT, 0, stream>>>(dd2, Wd3, bd3, out_y, G);
}
